// NCA_61177514164658
// MI455X (gfx1250) — compile-verified, matrix operand order checked by reading, not run on hardware
//
#include <hip/hip_runtime.h>
#include <hip/hip_bf16.h>
#include <math.h>

// ---------------------------------------------------------------------------
// Types for CDNA5 WMMA (wave32, 16x16x32 bf16 -> f32 accumulate) + TDM groups
// ---------------------------------------------------------------------------
typedef __attribute__((ext_vector_type(16))) __bf16 v16bf;
typedef __attribute__((ext_vector_type(8)))  __bf16 bf16x8;
typedef __attribute__((ext_vector_type(8)))  float  v8f;
typedef __attribute__((ext_vector_type(4)))  unsigned int u32x4;
typedef __attribute__((ext_vector_type(8)))  int          i32x8;
typedef __attribute__((ext_vector_type(4)))  int          i32x4;

#define LN_EPS 1e-5f

static constexpr int    B_     = 256;
static constexpr int    C_     = 256;
static constexpr int    HW_    = 225;   // 15*15
static constexpr int    STEPS_ = 8;
static constexpr size_t NXE    = (size_t)B_ * C_ * HW_;       // 14,745,600
static constexpr size_t CATE   = (size_t)B_ * 4 * C_ * HW_;   // 58,982,400
static constexpr int    PERB   = C_ * HW_;                    // 57,600

// ---------------------------------------------------------------------------
// fp32 -> bf16 conversion
// ---------------------------------------------------------------------------
__global__ __launch_bounds__(256) void f32_to_bf16_k(const float* __restrict__ in,
                                                     __bf16* __restrict__ out,
                                                     size_t n) {
    size_t i = (size_t)blockIdx.x * blockDim.x + threadIdx.x;
    if (i < n) out[i] = (__bf16)in[i];
}

// ---------------------------------------------------------------------------
// Tensor Data Mover: 2D tile (32 rows x 16 cols, 2B elems) global -> LDS.
// D# layout per CDNA5 ISA:
//   group0: [1:0]=count=1, [63:32]=lds_addr, [120:64]=global_addr, [127:126]=2
//   group1: [17:16]=data_size(1 => 2B), [79:48]=tensor_dim0 (remaining x; OOB
//           reads beyond it return zero -> hardware pad of the ragged tile),
//           [111:80]=tensor_dim1, [127:112]=tile_dim0=16, [143:128]=tile_dim1=32,
//           [207:160]=tensor_dim0_stride
// Arity differs across toolchains (bridge doc): 5 args on ROCm7.2/clang-22,
// 6 args on amdgpu-toolchain/clang-23.
// ---------------------------------------------------------------------------
__device__ __forceinline__ void tdm_load_tile_to_lds(const __bf16* gsrc,
                                                     unsigned lds_off,
                                                     int remX, int remK,
                                                     int strideX) {
    unsigned long long ga = (unsigned long long)(uintptr_t)gsrc;
    u32x4 g0 = { 1u,                                   // count=1, user mode
                 lds_off,                              // LDS byte address
                 (unsigned)(ga & 0xFFFFFFFFu),
                 (unsigned)((ga >> 32) & 0x01FFFFFFu) | 0x80000000u }; // type=2
    i32x8 g1 = { (int)(1u << 16),                                       // data_size=2B
                 (int)(((unsigned)remX & 0xFFFFu) << 16),               // dim0 lo16
                 (int)((((unsigned)remX >> 16) & 0xFFFFu) |
                       (((unsigned)remK & 0xFFFFu) << 16)),             // dim0 hi | dim1 lo
                 (int)((((unsigned)remK >> 16) & 0xFFFFu) | (16u << 16)), // dim1 hi | tile_dim0
                 32,                                                     // tile_dim1
                 strideX,                                                // dim0 stride lo32
                 0, 0 };
    i32x4 z4 = { 0, 0, 0, 0 };
#if __clang_major__ >= 23
    i32x8 z8 = { 0, 0, 0, 0, 0, 0, 0, 0 };
    __builtin_amdgcn_tensor_load_to_lds(g0, g1, z4, z4, z8, 0);
#else
    __builtin_amdgcn_tensor_load_to_lds(g0, g1, z4, z4, 0);
#endif
}

// ---------------------------------------------------------------------------
// Implicit-GEMM convolution via v_wmma_f32_16x16x32_bf16 with LDS-shared B.
//   Workgroup (8 waves) <-> (batch b, pixel tile of 16). Each wave computes
//   2 cout tiles (wave, wave+8) so the workgroup covers all 256 channels and
//   the im2col B block for each 32-wide K chunk is staged ONCE into LDS.
//   TAPS==1 (1x1 convs): B block is a dense 2D tile -> staged by the TDM,
//   which also zero-fills the ragged last pixel tile via tensor_dim0.
//   TAPS>1: cooperative gather staging (n-major layout -> ds_load_b128 reads).
// ---------------------------------------------------------------------------
template<int KH, int KW, int PAD, int DIL, int CIN, bool DO_ELU, bool OUT_BF16>
__global__ __launch_bounds__(256) void conv_wmma_k(const __bf16* __restrict__ xin,
                                                   const __bf16* __restrict__ wgt,
                                                   const float*  __restrict__ bias,
                                                   void* __restrict__ outp,
                                                   int outCoff, int outCstride) {
    constexpr int  TAPS    = KH * KW;
    constexpr int  K       = CIN * TAPS;
    constexpr bool USE_TDM = (TAPS == 1);

    extern __shared__ char smemRaw[];
    __bf16* Bs = (__bf16*)smemRaw;   // manual: [n=16][k=32]; TDM: [k=32][n=16]

    const int tid  = threadIdx.x;
    const int lane = tid & 31;
    const int wave = tid >> 5;
    const int b    = blockIdx.x / 15;
    const int pixT = blockIdx.x - b * 15;

    const int lh     = lane >> 4;       // K half select (A/B 16-bit layout)
    const int ln16   = lane & 15;       // M row / N col
    const int pix    = pixT * 16 + ln16;
    const int coutT0 = wave;
    const int coutT1 = wave + 8;

    const __bf16* __restrict__ ximg  = xin + (size_t)b * CIN * HW_;
    const __bf16* __restrict__ arow0 = wgt + (size_t)(coutT0 * 16 + ln16) * K + lh * 8;
    const __bf16* __restrict__ arow1 = wgt + (size_t)(coutT1 * 16 + ln16) * K + lh * 8;

    v8f acc0 = {};
    v8f acc1 = {};

    for (int kb = 0; kb < K; kb += 32) {
        // ---- stage B block [32 k x 16 n] into LDS ----
        if (USE_TDM) {
            if (wave == 0) {
                tdm_load_tile_to_lds(ximg + (size_t)kb * HW_ + pixT * 16, 0u,
                                     HW_ - pixT * 16, CIN - kb, HW_);
                __builtin_amdgcn_s_wait_tensorcnt(0);
            }
        } else {
            for (int e = tid; e < 512; e += 256) {
                int n    = e >> 5;
                int kloc = e & 31;
                int kk   = kb + kloc;
                int cin  = kk / TAPS;
                int tap  = kk - cin * TAPS;
                int dh   = tap / KW;
                int dw   = tap - dh * KW;
                int p    = pixT * 16 + n;
                __bf16 v = (__bf16)0.0f;
                if (p < HW_) {
                    int oh = p / 15;
                    int ow = p - oh * 15;
                    int ih = oh - PAD + dh * DIL;
                    int iw = ow - PAD + dw * DIL;
                    if ((unsigned)ih < 15u && (unsigned)iw < 15u)
                        v = ximg[(size_t)cin * HW_ + ih * 15 + iw];
                }
                Bs[n * 32 + kloc] = v;
            }
        }
        __syncthreads();

        // prefetch next weight chunk (streams linearly, L2-resident)
        if (kb + 32 < K) {
            __builtin_prefetch((const void*)(arow0 + kb + 32), 0, 3);
            __builtin_prefetch((const void*)(arow1 + kb + 32), 0, 3);
        }

        // ---- B fragment from LDS ----
        v16bf Bv;
        if (USE_TDM) {
            // TDM deposited k-major rows of 16 pixels
#pragma unroll
            for (int e = 0; e < 16; ++e) {
                int kloc = ((e >> 3) << 4) + (lh << 3) + (e & 7);
                Bv[e] = Bs[kloc * 16 + ln16];
            }
        } else {
            const __bf16* bn = Bs + ln16 * 32 + lh * 8;
            bf16x8 blo = *reinterpret_cast<const bf16x8*>(bn);
            bf16x8 bhi = *reinterpret_cast<const bf16x8*>(bn + 16);
            Bv = __builtin_shufflevector(blo, bhi,
                     0, 1, 2, 3, 4, 5, 6, 7, 8, 9, 10, 11, 12, 13, 14, 15);
        }

        // ---- A fragments (two 16B loads each) + 2x WMMA ----
        {
            bf16x8 l0 = *reinterpret_cast<const bf16x8*>(arow0 + kb);
            bf16x8 h0 = *reinterpret_cast<const bf16x8*>(arow0 + kb + 16);
            v16bf  A0 = __builtin_shufflevector(l0, h0,
                            0, 1, 2, 3, 4, 5, 6, 7, 8, 9, 10, 11, 12, 13, 14, 15);
            acc0 = __builtin_amdgcn_wmma_f32_16x16x32_bf16(
                       false, A0, false, Bv, (short)0, acc0, false, false);

            bf16x8 l1 = *reinterpret_cast<const bf16x8*>(arow1 + kb);
            bf16x8 h1 = *reinterpret_cast<const bf16x8*>(arow1 + kb + 16);
            v16bf  A1 = __builtin_shufflevector(l1, h1,
                            0, 1, 2, 3, 4, 5, 6, 7, 8, 9, 10, 11, 12, 13, 14, 15);
            acc1 = __builtin_amdgcn_wmma_f32_16x16x32_bf16(
                       false, A1, false, Bv, (short)0, acc1, false, false);
        }
        __syncthreads();
    }

    // ---- epilogue: bias (+ELU). Lane ln16 = column N; acc VGPR rr -> M ----
    if (pix < HW_) {
#pragma unroll
        for (int rr = 0; rr < 8; ++rr) {
            int M = rr + lh * 8;
            {
                int   cout = coutT0 * 16 + M;
                float v    = acc0[rr] + bias[cout];
                if (DO_ELU) v = (v > 0.0f) ? v : expm1f(v);
                size_t oi = ((size_t)b * outCstride + outCoff + cout) * HW_ + pix;
                if (OUT_BF16) ((__bf16*)outp)[oi] = (__bf16)v;
                else          ((float*)outp)[oi]  = v;
            }
            {
                int   cout = coutT1 * 16 + M;
                float v    = acc1[rr] + bias[cout];
                if (DO_ELU) v = (v > 0.0f) ? v : expm1f(v);
                size_t oi = ((size_t)b * outCstride + outCoff + cout) * HW_ + pix;
                if (OUT_BF16) ((__bf16*)outp)[oi] = (__bf16)v;
                else          ((float*)outp)[oi]  = v;
            }
        }
    }
}

// ---------------------------------------------------------------------------
// LayerNorm stats over (C,H,W) per batch: mu, rsqrt(var+eps)  (biased var)
// ---------------------------------------------------------------------------
__global__ __launch_bounds__(256) void ln_stats_k(const float* __restrict__ y,
                                                  float* __restrict__ mu,
                                                  float* __restrict__ rstd) {
    const int b = blockIdx.x;
    const float* p = y + (size_t)b * PERB;
    float s = 0.0f, s2 = 0.0f;
    for (int i = threadIdx.x; i < PERB; i += 256) {
        float v = p[i];
        s += v; s2 += v * v;
    }
    __shared__ float sh[256], sh2[256];
    sh[threadIdx.x] = s; sh2[threadIdx.x] = s2;
    __syncthreads();
    for (int o = 128; o > 0; o >>= 1) {
        if (threadIdx.x < o) {
            sh[threadIdx.x]  += sh[threadIdx.x + o];
            sh2[threadIdx.x] += sh2[threadIdx.x + o];
        }
        __syncthreads();
    }
    if (threadIdx.x == 0) {
        float m   = sh[0] * (1.0f / PERB);
        float var = sh2[0] * (1.0f / PERB) - m * m;
        mu[b]   = m;
        rstd[b] = rsqrtf(var + LN_EPS);
    }
}

// ---------------------------------------------------------------------------
// dx = ELU(LN(y)); also unbiased variance of dx per batch -> nca_var[b][s]
// ---------------------------------------------------------------------------
__global__ __launch_bounds__(256) void dx_var_k(const float* __restrict__ y,
                                                const float* __restrict__ mu,
                                                const float* __restrict__ rstd,
                                                const float* __restrict__ lnw,
                                                const float* __restrict__ lnb,
                                                float* __restrict__ dx,
                                                float* __restrict__ varOut,
                                                int s) {
    const int b = blockIdx.x;
    const float* p = y + (size_t)b * PERB;
    float* q = dx + (size_t)b * PERB;
    const float m = mu[b], rs = rstd[b];
    float sum = 0.0f, sum2 = 0.0f;
    for (int i = threadIdx.x; i < PERB; i += 256) {
        float v = (p[i] - m) * rs * lnw[i] + lnb[i];
        v = (v > 0.0f) ? v : expm1f(v);
        q[i] = v;
        sum += v; sum2 += v * v;
    }
    __shared__ float sh[256], sh2[256];
    sh[threadIdx.x] = sum; sh2[threadIdx.x] = sum2;
    __syncthreads();
    for (int o = 128; o > 0; o >>= 1) {
        if (threadIdx.x < o) {
            sh[threadIdx.x]  += sh[threadIdx.x + o];
            sh2[threadIdx.x] += sh2[threadIdx.x + o];
        }
        __syncthreads();
    }
    if (threadIdx.x == 0) {
        float n = (float)PERB;
        varOut[b * STEPS_ + s] = (sh2[0] - sh[0] * sh[0] / n) / (n - 1.0f);
    }
}

// ---------------------------------------------------------------------------
// Stochastic update: x = x*(1+rw) + dx*sp + U(0,1)*prob*ss
// (hash-based uniform; JAX threefry stream is not reproducible on-device)
// ---------------------------------------------------------------------------
__device__ __forceinline__ float unoise(unsigned i, unsigned s) {
    unsigned h = i * 2654435761u ^ (s * 40503u + 0x9E3779B9u);
    h ^= h >> 16; h *= 0x7feb352du;
    h ^= h >> 15; h *= 0x846ca68bu;
    h ^= h >> 16;
    return (float)(h >> 8) * (1.0f / 16777216.0f);
}

__global__ __launch_bounds__(256) void update_k(float* __restrict__ xc,
                                                const float* __restrict__ dx,
                                                const float* __restrict__ sp,
                                                const float* __restrict__ ss,
                                                const float* __restrict__ rw,
                                                const float* __restrict__ prob,
                                                int s, size_t n) {
    size_t i = (size_t)blockIdx.x * blockDim.x + threadIdx.x;
    if (i >= n) return;
    float a = sp[s];
    float c = ss[s] * prob[s];
    float w = 1.0f + rw[s];
    xc[i] = xc[i] * w + dx[i] * a + unoise((unsigned)i, (unsigned)s) * c;
}

// ---------------------------------------------------------------------------
// Host-side launch
// ---------------------------------------------------------------------------
extern "C" void kernel_launch(void* const* d_in, const int* in_sizes, int n_in,
                              void* d_out, int out_size, void* d_ws, size_t ws_size,
                              hipStream_t stream) {
    const float* x    = (const float*)d_in[0];
    const float* prob = (const float*)d_in[1];
    const float* w3   = (const float*)d_in[2];
    const float* b3   = (const float*)d_in[3];
    const float* w3d  = (const float*)d_in[4];
    const float* b3d  = (const float*)d_in[5];
    const float* w5   = (const float*)d_in[6];
    const float* b5   = (const float*)d_in[7];
    const float* w7   = (const float*)d_in[8];
    const float* b7   = (const float*)d_in[9];
    const float* wf   = (const float*)d_in[10];
    const float* bfc  = (const float*)d_in[11];
    const float* w1   = (const float*)d_in[12];
    const float* b1   = (const float*)d_in[13];
    const float* lnw  = (const float*)d_in[14];
    const float* lnb  = (const float*)d_in[15];
    const float* sp   = (const float*)d_in[16];
    const float* ss   = (const float*)d_in[17];
    const float* rw   = (const float*)d_in[18];

    float* outF   = (float*)d_out;
    float* xc     = outF;           // fp32 state lives in d_out (first section)
    float* varOut = outF + NXE;     // nca_var [B, 8]

    // ---- workspace carve-out ----
    char*  wsBase = (char*)d_ws;
    size_t off = 0;
    auto carve = [&](size_t bytes) -> void* {
        void* p = wsBase + off;
        off = (off + bytes + 255) & ~(size_t)255;
        return p;
    };
    __bf16* xbf   = (__bf16*)carve(NXE * 2);
    __bf16* xcbf  = (__bf16*)carve(NXE * 2);
    __bf16* catb  = (__bf16*)carve(CATE * 2);
    float*  ybuf  = (float*)carve(NXE * 4);
    float*  dxbuf = (float*)carve(NXE * 4);
    float*  mu    = (float*)carve(B_ * 4);
    float*  rstd  = (float*)carve(B_ * 4);
    __bf16* w3b   = (__bf16*)carve((size_t)256 * 2304 * 2);
    __bf16* w3db  = (__bf16*)carve((size_t)256 * 2304 * 2);
    __bf16* w5b   = (__bf16*)carve((size_t)256 * 6400 * 2);
    __bf16* w7b   = (__bf16*)carve((size_t)256 * 12544 * 2);
    __bf16* wfb   = (__bf16*)carve((size_t)256 * 1024 * 2);
    __bf16* w1b   = (__bf16*)carve((size_t)256 * 256 * 2);

    auto cvt = [&](const float* src, __bf16* dst, size_t n) {
        f32_to_bf16_k<<<(unsigned)((n + 255) / 256), 256, 0, stream>>>(src, dst, n);
    };

    // ---- precision lowering: activations + weights to bf16 ----
    cvt(x,   xbf,  NXE);
    cvt(w3,  w3b,  (size_t)256 * 2304);
    cvt(w3d, w3db, (size_t)256 * 2304);
    cvt(w5,  w5b,  (size_t)256 * 6400);
    cvt(w7,  w7b,  (size_t)256 * 12544);
    cvt(wf,  wfb,  (size_t)256 * 1024);
    cvt(w1,  w1b,  (size_t)256 * 256);

    const int    convBlocks = B_ * 15;      // one WG per (batch, pixel-tile)
    const size_t convShmem  = 16 * 32 * sizeof(__bf16);   // 1 KB B-block

    // ---- multi-scale perception -> concat buffer (bf16, ELU fused) ----
    conv_wmma_k<3,3,1,1,256,true,true><<<convBlocks, 256, convShmem, stream>>>(xbf, w3b,  b3,  catb, 0,    1024);
    conv_wmma_k<3,3,2,2,256,true,true><<<convBlocks, 256, convShmem, stream>>>(xbf, w3db, b3d, catb, 256,  1024);
    conv_wmma_k<5,5,2,1,256,true,true><<<convBlocks, 256, convShmem, stream>>>(xbf, w5b,  b5,  catb, 512,  1024);
    conv_wmma_k<7,7,3,1,256,true,true><<<convBlocks, 256, convShmem, stream>>>(xbf, w7b,  b7,  catb, 768,  1024);

    // ---- 1x1 fusion (K=1024, TDM-staged B) -> fp32 state in d_out ----
    conv_wmma_k<1,1,0,1,1024,true,false><<<convBlocks, 256, convShmem, stream>>>(catb, wfb, bfc, xc, 0, 256);

    // ---- 8 NCA steps ----
    const unsigned elemBlocks = (unsigned)((NXE + 255) / 256);
    for (int s = 0; s < STEPS_; ++s) {
        cvt(xc, xcbf, NXE);
        conv_wmma_k<1,1,0,1,256,false,false><<<convBlocks, 256, convShmem, stream>>>(xcbf, w1b, b1, ybuf, 0, 256);
        ln_stats_k<<<B_, 256, 0, stream>>>(ybuf, mu, rstd);
        dx_var_k<<<B_, 256, 0, stream>>>(ybuf, mu, rstd, lnw, lnb, dxbuf, varOut, s);
        update_k<<<elemBlocks, 256, 0, stream>>>(xc, dxbuf, sp, ss, rw, prob, s, NXE);
    }
}
